// MultiTaskPointNet2Transformer_50714973831841
// MI455X (gfx1250) — compile-verified, matrix operand order checked by reading, not run on hardware
//
#include <hip/hip_runtime.h>
#include <hip/hip_bf16.h>

typedef __attribute__((ext_vector_type(16))) _Float16 v16h;
typedef __attribute__((ext_vector_type(8)))  float    v8f;
typedef float f4u __attribute__((ext_vector_type(4), aligned(4)));

#define KNN_K 32
#define EPSBN 1e-5f

static inline int nblk(long long n) { return (int)((n + 255) / 256); }

// ---------------------------------------------------------------------------
// Transpose input x [B,6,N] -> pts0 [B,N,6] and xyz0 [B,N,3]
// ---------------------------------------------------------------------------
__global__ void k_transp(const float* __restrict__ x, int B, int Cc, int N,
                         float* __restrict__ pts, float* __restrict__ xyz) {
  int t = blockIdx.x * blockDim.x + threadIdx.x;
  if (t >= B * N) return;
  int b = t / N, n = t % N;
  for (int c = 0; c < Cc; ++c) {
    float v = x[((size_t)b * Cc + c) * N + n];
    pts[(size_t)t * Cc + c] = v;
    if (c < 3) xyz[(size_t)t * 3 + c] = v;
  }
}

// ---------------------------------------------------------------------------
// Farthest point sampling: one block per batch, iterative argmax reduction
// ---------------------------------------------------------------------------
__global__ void k_fps(const float* __restrict__ xyz, int N, int S,
                      int* __restrict__ out, float* __restrict__ distws) {
  int b = blockIdx.x;
  int tid = threadIdx.x;
  const float* P = xyz + (size_t)b * N * 3;
  float* D = distws + (size_t)b * N;
  __shared__ float best[256];
  __shared__ int besti[256];
  __shared__ float fx, fy, fz;
  __shared__ int cur;
  for (int i = tid; i < N; i += 256) D[i] = 1e10f;
  if (tid == 0) cur = 0;
  __syncthreads();
  for (int s = 0; s < S; ++s) {
    if (tid == 0) {
      out[b * S + s] = cur;
      fx = P[cur * 3 + 0]; fy = P[cur * 3 + 1]; fz = P[cur * 3 + 2];
    }
    __syncthreads();
    float bm = -1.f; int bi = 0;
    for (int i = tid; i < N; i += 256) {
      float dx = P[i * 3 + 0] - fx, dy = P[i * 3 + 1] - fy, dz = P[i * 3 + 2] - fz;
      float d = dx * dx + dy * dy + dz * dz;
      float dd = fminf(D[i], d);
      D[i] = dd;
      if (dd > bm) { bm = dd; bi = i; }
    }
    best[tid] = bm; besti[tid] = bi;
    __syncthreads();
    for (int off = 128; off > 0; off >>= 1) {
      if (tid < off && best[tid + off] > best[tid]) {
        best[tid] = best[tid + off]; besti[tid] = besti[tid + off];
      }
      __syncthreads();
    }
    if (tid == 0) cur = besti[0];
    __syncthreads();
  }
}

// gather centroid coords: out[b,s,:] = xyz[b, idx[b,s], :]
__global__ void k_gather3(const float* __restrict__ xyz, int N,
                          const int* __restrict__ idx, int S,
                          float* __restrict__ out, int total) {
  int t = blockIdx.x * blockDim.x + threadIdx.x;
  if (t >= total) return;
  int b = t / S;
  int pi = idx[t];
  const float* p = xyz + ((size_t)b * N + pi) * 3;
  out[t * 3 + 0] = p[0]; out[t * 3 + 1] = p[1]; out[t * 3 + 2] = p[2];
}

// ---------------------------------------------------------------------------
// kNN top-32: one wave (32 lanes) per query; per-lane local top-32 lists in
// LDS over a strided scan, then 32-round merge.
// ---------------------------------------------------------------------------
__global__ void k_knn(const float* __restrict__ q, const float* __restrict__ p,
                      int S, int N, int* __restrict__ out) {
  int b = blockIdx.x / S;
  int s = blockIdx.x % S;
  int lane = threadIdx.x;
  __shared__ float ld[32][33];
  __shared__ int   li[32][33];
  __shared__ float hd[32];
  __shared__ int   hi[32];
  __shared__ int   hp[32];
  float qx = q[(b * S + s) * 3 + 0];
  float qy = q[(b * S + s) * 3 + 1];
  float qz = q[(b * S + s) * 3 + 2];
  for (int j = 0; j < 32; ++j) { ld[lane][j] = 3e38f; li[lane][j] = 0; }
  float worst = 3e38f; int wpos = 0;
  const float* P = p + (size_t)b * N * 3;
  for (int i = lane; i < N; i += 32) {
    float dx = P[i * 3 + 0] - qx, dy = P[i * 3 + 1] - qy, dz = P[i * 3 + 2] - qz;
    float d = dx * dx + dy * dy + dz * dz;
    if (d < worst) {
      ld[lane][wpos] = d; li[lane][wpos] = i;
      worst = -1.f;
      for (int j = 0; j < 32; ++j)
        if (ld[lane][j] > worst) { worst = ld[lane][j]; wpos = j; }
    }
  }
  __syncthreads();
  for (int r = 0; r < KNN_K; ++r) {
    float bm = 3e38f; int bp = 0;
    for (int j = 0; j < 32; ++j)
      if (ld[lane][j] < bm) { bm = ld[lane][j]; bp = j; }
    hd[lane] = bm; hi[lane] = li[lane][bp]; hp[lane] = bp;
    __syncthreads();
    if (lane == 0) {
      float gm = 3e38f; int gl = 0;
      for (int l = 0; l < 32; ++l)
        if (hd[l] < gm) { gm = hd[l]; gl = l; }
      out[((size_t)(b * S + s)) * KNN_K + r] = hi[gl];
      ld[gl][hp[gl]] = 3e38f;
    }
    __syncthreads();
  }
}

// ---------------------------------------------------------------------------
// Build grouped features [B*S*K, 3+Cf]: local coords + gathered point feats
// ---------------------------------------------------------------------------
__global__ void k_group(const float* __restrict__ xyz, const float* __restrict__ nxyz,
                        const int* __restrict__ nidx, const float* __restrict__ feats,
                        int N, int S, int Cf, float* __restrict__ out, int B) {
  int t = blockIdx.x * blockDim.x + threadIdx.x;
  int total = B * S * KNN_K;
  if (t >= total) return;
  int s = (t / KNN_K) % S;
  int b = t / (KNN_K * S);
  int pi = nidx[t];
  int C = 3 + Cf;
  float* o = out + (size_t)t * C;
  const float* px = xyz + ((size_t)b * N + pi) * 3;
  const float* cx = nxyz + ((size_t)(b * S + s)) * 3;
  o[0] = px[0] - cx[0]; o[1] = px[1] - cx[1]; o[2] = px[2] - cx[2];
  const float* f = feats + ((size_t)b * N + pi) * Cf;
  for (int c = 0; c < Cf; ++c) o[3 + c] = f[c];
}

// ---------------------------------------------------------------------------
// WMMA fragment helpers (ISA 7.12.2 layout: lanes 0-15 -> K {0..7,16..23},
// lanes 16-31 -> K {8..15,24..31}; caller's pointer already includes half*8).
// ---------------------------------------------------------------------------
__device__ __forceinline__ void ldraw(const float* __restrict__ p, f4u* __restrict__ r) {
  r[0] = *(const f4u*)(p);
  r[1] = *(const f4u*)(p + 4);
  r[2] = *(const f4u*)(p + 16);
  r[3] = *(const f4u*)(p + 20);
}

__device__ __forceinline__ v16h cvtfrag(const f4u* __restrict__ r) {
  v16h o;
#pragma unroll
  for (int i = 0; i < 4; ++i) {
    o[i]      = (_Float16)r[0][i];
    o[i + 4]  = (_Float16)r[1][i];
    o[i + 8]  = (_Float16)r[2][i];
    o[i + 12] = (_Float16)r[3][i];
  }
  return o;
}

__device__ __forceinline__ v16h ld_frag_strided(const float* __restrict__ p,
                                                int kb, int stride) {
  v16h r;
#pragma unroll
  for (int i = 0; i < 8; ++i) {
    r[i]     = (_Float16)p[(size_t)(kb + i) * stride];
    r[i + 8] = (_Float16)p[(size_t)(kb + 16 + i) * stride];
  }
  return r;
}

// branch-free masked tail loader (clamped address + cndmask-zero)
__device__ __forceinline__ v16h ld_frag_tail(const float* __restrict__ p,
                                             int kb, int stride, int K) {
  v16h r;
#pragma unroll
  for (int i = 0; i < 8; ++i) {
    int k1 = kb + i, k2 = kb + 16 + i;
    int c1 = k1 < K ? k1 : K - 1;
    int c2 = k2 < K ? k2 : K - 1;
    float a1 = p[(size_t)c1 * stride];
    float a2 = p[(size_t)c2 * stride];
    r[i]     = (_Float16)(k1 < K ? a1 : 0.f);
    r[i + 8] = (_Float16)(k2 < K ? a2 : 0.f);
  }
  return r;
}

// ---------------------------------------------------------------------------
// WMMA GEMM: C[m,n] = alpha * sum_k A(m,k)*B(n,k) + bias[n]
//   A(m,k) = A[m*lda + k]  (f32 -> f16 on load); B(n,k) = W[n*ldwr + k*ldwc]
// Each wave computes a 32 x (16*NT) tile: 2 A row-fragments reuse NT B
// fragments (8 wmma per 32-wide k-step at NT=4).  Software-pipelined:
// raw loads for step k+1 issue between conversion and the wmma burst of
// step k, so WMMA latency hides VMEM latency (single wait per step).
// Requires Nn % (16*NT) == 0 (host dispatch). M rows clamped, stores masked.
// Batched via gridDim.y with two-level offsets (bi/H, bi%H).
// ---------------------------------------------------------------------------
template <int NT, bool BCONTIG>
__global__ void k_gemm(const float* __restrict__ A, const float* __restrict__ W,
                       const float* __restrict__ bias, float* __restrict__ C,
                       int M, int Nn, int K, int lda, int ldwr, int ldwc, int ldc,
                       int H, long long sA0, long long sA1, long long sB0, long long sB1,
                       long long sC0, long long sC1, float alpha) {
  int bi = blockIdx.y;
  const float* Ab = A + (long long)(bi / H) * sA0 + (long long)(bi % H) * sA1;
  const float* Wb = W + (long long)(bi / H) * sB0 + (long long)(bi % H) * sB1;
  float* Cb = C + (long long)(bi / H) * sC0 + (long long)(bi % H) * sC1;

  int wave = blockIdx.x * (blockDim.x >> 5) + (threadIdx.x >> 5);
  int mt_cnt = (M + 31) >> 5;
  int ng_cnt = Nn / (16 * NT);
  if (wave >= mt_cnt * ng_cnt) return;
  int mt = wave / ng_cnt, ng = wave % ng_cnt;

  int lane = threadIdx.x & 31;
  int half = lane >> 4;
  int lid  = lane & 15;
  int kb0  = half * 8;

  int am0 = mt * 32 + lid;        if (am0 >= M) am0 = M - 1;   // clamp
  int am1 = mt * 32 + 16 + lid;   if (am1 >= M) am1 = M - 1;
  const float* arow0 = Ab + (size_t)am0 * lda + kb0;
  const float* arow1 = Ab + (size_t)am1 * lda + kb0;
  const float* brow[NT];
#pragma unroll
  for (int t = 0; t < NT; ++t) {
    int bn = ng * (16 * NT) + t * 16 + lid;      // always < Nn
    brow[t] = Wb + (size_t)bn * ldwr + (BCONTIG ? kb0 : 0);
  }

  v8f acc[2][NT];
#pragma unroll
  for (int t = 0; t < NT; ++t) { acc[0][t] = (v8f){}; acc[1][t] = (v8f){}; }

  int kmain = K & ~31;
  f4u araw0[4], araw1[4], braw[NT][4];
  if (kmain > 0) {
    ldraw(arow0, araw0);
    ldraw(arow1, araw1);
    if (BCONTIG) {
#pragma unroll
      for (int t = 0; t < NT; ++t) ldraw(brow[t], braw[t]);
    }
    for (int k0 = 0; k0 < kmain; k0 += 32) {
      v16h af0 = cvtfrag(araw0);
      v16h af1 = cvtfrag(araw1);
      v16h bf[NT];
#pragma unroll
      for (int t = 0; t < NT; ++t)
        bf[t] = BCONTIG ? cvtfrag(braw[t]) : ld_frag_strided(brow[t], k0 + kb0, ldwc);
      int kn = k0 + 32;
      if (kn < kmain) {                      // prefetch next step into raw regs
        ldraw(arow0 + kn, araw0);
        ldraw(arow1 + kn, araw1);
        if (BCONTIG) {
#pragma unroll
          for (int t = 0; t < NT; ++t) ldraw(brow[t] + kn, braw[t]);
        }
      }
#pragma unroll
      for (int t = 0; t < NT; ++t)
        acc[0][t] = __builtin_amdgcn_wmma_f32_16x16x32_f16(false, af0, false, bf[t],
                                                           (short)0, acc[0][t], false, false);
#pragma unroll
      for (int t = 0; t < NT; ++t)
        acc[1][t] = __builtin_amdgcn_wmma_f32_16x16x32_f16(false, af1, false, bf[t],
                                                           (short)0, acc[1][t], false, false);
    }
  }
  if (kmain < K) {
    v16h af0 = ld_frag_tail(Ab + (size_t)am0 * lda, kmain + kb0, 1, K);
    v16h af1 = ld_frag_tail(Ab + (size_t)am1 * lda, kmain + kb0, 1, K);
#pragma unroll
    for (int t = 0; t < NT; ++t) {
      const float* wb = Wb + (size_t)(ng * (16 * NT) + t * 16 + lid) * ldwr;
      v16h bf = ld_frag_tail(wb, kmain + kb0, BCONTIG ? 1 : ldwc, K);
      acc[0][t] = __builtin_amdgcn_wmma_f32_16x16x32_f16(false, af0, false, bf,
                                                         (short)0, acc[0][t], false, false);
      acc[1][t] = __builtin_amdgcn_wmma_f32_16x16x32_f16(false, af1, false, bf,
                                                         (short)0, acc[1][t], false, false);
    }
  }

#pragma unroll
  for (int h2 = 0; h2 < 2; ++h2) {
    int mb = mt * 32 + h2 * 16 + half * 8;
#pragma unroll
    for (int t = 0; t < NT; ++t) {
      int bn = ng * (16 * NT) + t * 16 + lid;
      float bvv = bias ? bias[bn] : 0.f;
#pragma unroll
      for (int r = 0; r < 8; ++r) {
        int m = mb + r;
        if (m < M) Cb[(size_t)m * ldc + bn] = acc[h2][t][r] * alpha + bvv;
      }
    }
  }
}

// ---------------------------------------------------------------------------
// BatchNorm statistics: one block per channel -> scale[c], shift[c]
// ---------------------------------------------------------------------------
__global__ void k_bnstats(const float* __restrict__ X, int M, int C,
                          const float* __restrict__ g, const float* __restrict__ bt,
                          float* __restrict__ scale, float* __restrict__ shift) {
  int c = blockIdx.x;
  __shared__ float s1[256], s2[256];
  float a = 0.f, b2 = 0.f;
  for (int i = threadIdx.x; i < M; i += blockDim.x) {
    float v = X[(size_t)i * C + c];
    a += v; b2 += v * v;
  }
  s1[threadIdx.x] = a; s2[threadIdx.x] = b2;
  __syncthreads();
  for (int off = 128; off > 0; off >>= 1) {
    if (threadIdx.x < off) {
      s1[threadIdx.x] += s1[threadIdx.x + off];
      s2[threadIdx.x] += s2[threadIdx.x + off];
    }
    __syncthreads();
  }
  if (threadIdx.x == 0) {
    float mean = s1[0] / (float)M;
    float var  = s2[0] / (float)M - mean * mean;
    float sc = g[c] * rsqrtf(fmaxf(var, 0.f) + EPSBN);
    scale[c] = sc;
    shift[c] = bt[c] - mean * sc;
  }
}

__global__ void k_bnrelu(float* __restrict__ X, long long total, int C,
                         const float* __restrict__ scale, const float* __restrict__ shift) {
  long long t = (long long)blockIdx.x * blockDim.x + threadIdx.x;
  if (t >= total) return;
  int c = (int)(t % C);
  float v = X[t] * scale[c] + shift[c];
  X[t] = v > 0.f ? v : 0.f;
}

// max over K: in [rows,K,C] -> out [rows,C]
__global__ void k_maxk(const float* __restrict__ in, int K, int C,
                       float* __restrict__ out, long long total) {
  long long t = (long long)blockIdx.x * blockDim.x + threadIdx.x;
  if (t >= total) return;
  int c = (int)(t % C);
  long long r = t / C;
  const float* p = in + (r * (long long)K) * C + c;
  float m = -3e38f;
  for (int k = 0; k < K; ++k) m = fmaxf(m, p[(long long)k * C]);
  out[t] = m;
}

// LayerNorm (per row), y separate from x
__global__ void k_ln(const float* __restrict__ x, int M, int C,
                     const float* __restrict__ g, const float* __restrict__ b,
                     float* __restrict__ y) {
  int m = blockIdx.x * blockDim.x + threadIdx.x;
  if (m >= M) return;
  const float* r = x + (size_t)m * C;
  float s = 0.f;
  for (int c = 0; c < C; ++c) s += r[c];
  float mean = s / (float)C;
  float v = 0.f;
  for (int c = 0; c < C; ++c) { float d = r[c] - mean; v += d * d; }
  float rs = rsqrtf(v / (float)C + EPSBN);
  float* o = y + (size_t)m * C;
  for (int c = 0; c < C; ++c) o[c] = g[c] * (r[c] - mean) * rs + b[c];
}

__global__ void k_softmax(float* __restrict__ x, int rows, int L) {
  int m = blockIdx.x * blockDim.x + threadIdx.x;
  if (m >= rows) return;
  float* r = x + (size_t)m * L;
  float mx = -3e38f;
  for (int i = 0; i < L; ++i) mx = fmaxf(mx, r[i]);
  float s = 0.f;
  for (int i = 0; i < L; ++i) { float e = expf(r[i] - mx); r[i] = e; s += e; }
  float inv = 1.f / s;
  for (int i = 0; i < L; ++i) r[i] *= inv;
}

__global__ void k_add(float* __restrict__ y, const float* __restrict__ x, long long n) {
  long long t = (long long)blockIdx.x * blockDim.x + threadIdx.x;
  if (t < n) y[t] += x[t];
}

__global__ void k_gelu(float* __restrict__ x, long long n) {
  long long t = (long long)blockIdx.x * blockDim.x + threadIdx.x;
  if (t < n) {
    float v = x[t];
    x[t] = 0.5f * v * (1.f + erff(v * 0.70710678118654752f));
  }
}

// copy columns into a concat buffer; src row = (m / mdiv) for broadcast
__global__ void k_copycols(float* __restrict__ dst, int dld, int doff,
                           const float* __restrict__ src, int sld, int mdiv,
                           int C, long long total) {
  long long t = (long long)blockIdx.x * blockDim.x + threadIdx.x;
  if (t >= total) return;
  int c = (int)(t % C);
  long long m = t / C;
  dst[m * (long long)dld + doff + c] = src[(m / mdiv) * (long long)sld + c];
}

// three-nearest-neighbor search + inverse-distance weights
__global__ void k_nn3(const float* __restrict__ xyz1, int N1,
                      const float* __restrict__ xyz2, int N2, int B,
                      int* __restrict__ idx, float* __restrict__ w) {
  int t = blockIdx.x * blockDim.x + threadIdx.x;
  if (t >= B * N1) return;
  int b = t / N1;
  float qx = xyz1[t * 3 + 0], qy = xyz1[t * 3 + 1], qz = xyz1[t * 3 + 2];
  float d0 = 3e38f, d1 = 3e38f, d2 = 3e38f;
  int i0 = 0, i1 = 0, i2 = 0;
  const float* P = xyz2 + (size_t)b * N2 * 3;
  for (int j = 0; j < N2; ++j) {
    float dx = P[j * 3 + 0] - qx, dy = P[j * 3 + 1] - qy, dz = P[j * 3 + 2] - qz;
    float d = dx * dx + dy * dy + dz * dz;
    if (d < d0)      { d2 = d1; i2 = i1; d1 = d0; i1 = i0; d0 = d; i0 = j; }
    else if (d < d1) { d2 = d1; i2 = i1; d1 = d;  i1 = j; }
    else if (d < d2) { d2 = d;  i2 = j; }
  }
  float w0 = 1.f / (d0 + 1e-8f), w1 = 1.f / (d1 + 1e-8f), w2 = 1.f / (d2 + 1e-8f);
  float ws = w0 + w1 + w2;
  idx[t * 3 + 0] = i0; idx[t * 3 + 1] = i1; idx[t * 3 + 2] = i2;
  w[t * 3 + 0] = w0 / ws; w[t * 3 + 1] = w1 / ws; w[t * 3 + 2] = w2 / ws;
}

// weighted gather into concat buffer columns
__global__ void k_interp(float* __restrict__ dst, int dld, int doff,
                         const float* __restrict__ feat, int C, int S2,
                         const int* __restrict__ idx, const float* __restrict__ w,
                         int N1, long long total) {
  long long t = (long long)blockIdx.x * blockDim.x + threadIdx.x;
  if (t >= total) return;
  int c = (int)(t % C);
  long long m = t / C;
  long long b = m / N1;
  const int* id = idx + m * 3;
  const float* ww = w + m * 3;
  const float* fb = feat + b * (long long)S2 * C;
  float s = ww[0] * fb[(long long)id[0] * C + c]
          + ww[1] * fb[(long long)id[1] * C + c]
          + ww[2] * fb[(long long)id[2] * C + c];
  dst[m * (long long)dld + doff + c] = s;
}

// final heads: out [B,O,N]
__global__ void k_head(const float* __restrict__ feat, int N, int C,
                       const float* __restrict__ W, const float* __restrict__ bias,
                       int O, float* __restrict__ out, int total) {
  int t = blockIdx.x * blockDim.x + threadIdx.x;
  if (t >= total) return;
  int n = t % N;
  int o = (t / N) % O;
  int b = t / (N * O);
  const float* f = feat + ((size_t)b * N + n) * C;
  const float* wr = W + (size_t)o * C;
  float s = bias[o];
  for (int c = 0; c < C; ++c) s += f[c] * wr[c];
  out[t] = s;
}

// ===========================================================================
extern "C" void kernel_launch(void* const* d_in, const int* in_sizes, int n_in,
                              void* d_out, int out_size, void* d_ws, size_t ws_size,
                              hipStream_t stream) {
  (void)in_sizes; (void)n_in; (void)out_size; (void)ws_size;
  const int Bz = 4, Np = 16384, S1 = 1024, S2 = 256, KK = KNN_K, D = 256, NH = 8, HD = 32;

  // ---- parse inputs (setup_inputs dict insertion order) ----
  int ip = 0;
  auto F = [&]() { return (const float*)d_in[ip++]; };
  struct Layer { const float *W, *b, *g, *bt; int cin, cout; };
  auto mk = [&](int ci, int co) {
    Layer L; L.W = F(); L.b = F(); L.g = F(); L.bt = F(); L.cin = ci; L.cout = co; return L;
  };
  const float* x = F();
  Layer sa1[3]  = { mk(9, 64),    mk(64, 64),   mk(64, 128) };
  Layer sa2[3]  = { mk(131, 128), mk(128, 128), mk(128, 256) };
  Layer sa3[3]  = { mk(259, 256), mk(256, 512), mk(512, 1024) };
  Layer l2pre   = mk(256, 256);
  struct Enc { const float *ln1g,*ln1b,*Wqkv,*bqkv,*Wo,*bo,*ln2g,*ln2b,*W1,*b1,*W2,*b2; };
  Enc tenc[2];
  for (int i = 0; i < 2; ++i) {
    tenc[i].ln1g = F(); tenc[i].ln1b = F();
    tenc[i].Wqkv = F(); tenc[i].bqkv = F();
    tenc[i].Wo = F();   tenc[i].bo = F();
    tenc[i].ln2g = F(); tenc[i].ln2b = F();
    tenc[i].W1 = F();   tenc[i].b1 = F();
    tenc[i].W2 = F();   tenc[i].b2 = F();
  }
  Layer l2post  = mk(256, 256);
  Layer fp3p[2] = { mk(1280, 256), mk(256, 256) };
  Layer fp2p[2] = { mk(384, 256),  mk(256, 128) };
  Layer fp1p[3] = { mk(134, 128),  mk(128, 128), mk(128, 128) };
  Layer sharedL = mk(128, 128);
  const float* semW = F(); const float* semb = F();
  const float* defW = F(); const float* defb = F();

  // ---- workspace bump allocator ----
  char* wp = (char*)d_ws;
  auto af = [&](size_t n) { float* r = (float*)wp; wp += ((n * sizeof(float) + 255) & ~(size_t)255); return r; };
  auto ai = [&](size_t n) { int*   r = (int*)wp;   wp += ((n * sizeof(int)   + 255) & ~(size_t)255); return r; };

  float* pts0   = af((size_t)Bz * Np * 6);
  float* xyz0   = af((size_t)Bz * Np * 3);
  float* fpsd   = af((size_t)Bz * Np);
  int*   idx1   = ai((size_t)Bz * S1);
  float* nxyz1  = af((size_t)Bz * S1 * 3);
  int*   nidx1  = ai((size_t)Bz * S1 * KK);
  float* l1_pts = af((size_t)Bz * S1 * 128);
  int*   idx2   = ai((size_t)Bz * S2);
  float* nxyz2  = af((size_t)Bz * S2 * 3);
  int*   nidx2  = ai((size_t)Bz * S2 * KK);
  float* l2_pts = af((size_t)Bz * S2 * D);
  float* xb     = af((size_t)Bz * S2 * D);
  float* hb     = af((size_t)Bz * S2 * D);
  float* qkv    = af((size_t)Bz * S2 * 3 * D);
  float* scores = af((size_t)Bz * NH * S2 * S2);
  float* ctx    = af((size_t)Bz * S2 * D);
  float* ffb    = af((size_t)Bz * S2 * 512);
  float* l3_pts = af((size_t)Bz * 1024);
  float* scl    = af(1024);
  float* shf    = af(1024);
  int*   idx3a  = ai((size_t)Bz * S1 * 3);
  float* w3a    = af((size_t)Bz * S1 * 3);
  int*   idx3b  = ai((size_t)Bz * Np * 3);
  float* w3b    = af((size_t)Bz * Np * 3);
  float* bufA   = af((size_t)Bz * S1 * KK * 128);   // 16.78M floats
  float* bufB   = af((size_t)Bz * S1 * KK * 128);

  // ---- launch helpers ----
  auto gemm = [&](const float* A, const float* W, const float* bias, float* C,
                  int M, int Nn, int K, int lda, int ldwr, int ldwc, int ldc,
                  int nb, int H, long long sA0, long long sA1,
                  long long sB0, long long sB1, long long sC0, long long sC1,
                  float alpha) {
    bool bc = (ldwc == 1);
    int nt = (Nn % 64 == 0) ? 4 : 2;   // all Nn are multiples of 32
    long long waves = (long long)((M + 31) / 32) * (Nn / (16 * nt));
    dim3 g((unsigned)((waves + 7) / 8), (unsigned)nb, 1);
    if (bc) {
      if (nt == 4) k_gemm<4, true><<<g, 256, 0, stream>>>(A, W, bias, C, M, Nn, K, lda, ldwr, ldwc, ldc, H, sA0, sA1, sB0, sB1, sC0, sC1, alpha);
      else         k_gemm<2, true><<<g, 256, 0, stream>>>(A, W, bias, C, M, Nn, K, lda, ldwr, ldwc, ldc, H, sA0, sA1, sB0, sB1, sC0, sC1, alpha);
    } else {
      if (nt == 4) k_gemm<4, false><<<g, 256, 0, stream>>>(A, W, bias, C, M, Nn, K, lda, ldwr, ldwc, ldc, H, sA0, sA1, sB0, sB1, sC0, sC1, alpha);
      else         k_gemm<2, false><<<g, 256, 0, stream>>>(A, W, bias, C, M, Nn, K, lda, ldwr, ldwc, ldc, H, sA0, sA1, sB0, sB1, sC0, sC1, alpha);
    }
  };
  auto gemm1 = [&](const float* A, const float* W, const float* bias, float* C,
                   int M, int Nn, int K) {
    gemm(A, W, bias, C, M, Nn, K, K, K, 1, Nn, 1, 1, 0, 0, 0, 0, 0, 0, 1.f);
  };
  auto run_mlp = [&](const float* in, int M, const Layer* Ls, int nl,
                     float* pA, float* pB) -> float* {
    const float* src = in;
    float* dst = pA;
    float* oth = pB;
    for (int l = 0; l < nl; ++l) {
      gemm1(src, Ls[l].W, Ls[l].b, dst, M, Ls[l].cout, Ls[l].cin);
      k_bnstats<<<Ls[l].cout, 256, 0, stream>>>(dst, M, Ls[l].cout, Ls[l].g, Ls[l].bt, scl, shf);
      long long tot = (long long)M * Ls[l].cout;
      k_bnrelu<<<nblk(tot), 256, 0, stream>>>(dst, tot, Ls[l].cout, scl, shf);
      src = dst;
      float* t = dst; dst = oth; oth = t;
    }
    return (float*)src;
  };

  // ---- stage 0: transpose input ----
  k_transp<<<nblk((long long)Bz * Np), 256, 0, stream>>>(x, Bz, 6, Np, pts0, xyz0);

  // ---- SA1 ----
  k_fps<<<Bz, 256, 0, stream>>>(xyz0, Np, S1, idx1, fpsd);
  k_gather3<<<nblk(Bz * S1), 256, 0, stream>>>(xyz0, Np, idx1, S1, nxyz1, Bz * S1);
  k_knn<<<Bz * S1, 32, 0, stream>>>(nxyz1, xyz0, S1, Np, nidx1);
  k_group<<<nblk((long long)Bz * S1 * KK), 256, 0, stream>>>(xyz0, nxyz1, nidx1, pts0, Np, S1, 6, bufB, Bz);
  {
    float* o1 = run_mlp(bufB, Bz * S1 * KK, sa1, 3, bufA, bufB);
    long long tot = (long long)Bz * S1 * 128;
    k_maxk<<<nblk(tot), 256, 0, stream>>>(o1, KK, 128, l1_pts, tot);
  }

  // ---- SA2 ----
  k_fps<<<Bz, 256, 0, stream>>>(nxyz1, S1, S2, idx2, fpsd);
  k_gather3<<<nblk(Bz * S2), 256, 0, stream>>>(nxyz1, S1, idx2, S2, nxyz2, Bz * S2);
  k_knn<<<Bz * S2, 32, 0, stream>>>(nxyz2, nxyz1, S2, S1, nidx2);
  k_group<<<nblk((long long)Bz * S2 * KK), 256, 0, stream>>>(nxyz1, nxyz2, nidx2, l1_pts, S1, S2, 128, bufB, Bz);
  {
    float* o2 = run_mlp(bufB, Bz * S2 * KK, sa2, 3, bufA, bufB);
    long long tot = (long long)Bz * S2 * 256;
    k_maxk<<<nblk(tot), 256, 0, stream>>>(o2, KK, 256, l2_pts, tot);
  }

  // ---- transformer bottleneck ----
  const int Mx = Bz * S2;  // 1024 tokens
  run_mlp(l2_pts, Mx, &l2pre, 1, xb, bufB);  // a -> xb
  for (int li = 0; li < 2; ++li) {
    Enc& e = tenc[li];
    k_ln<<<nblk(Mx), 256, 0, stream>>>(xb, Mx, D, e.ln1g, e.ln1b, hb);
    gemm1(hb, e.Wqkv, e.bqkv, qkv, Mx, 3 * D, D);
    // scores[b,h] = Q K^T / sqrt(hd)     (32 batches)
    gemm(qkv, qkv + D, nullptr, scores, S2, S2, HD, 3 * D, 3 * D, 1, S2,
         Bz * NH, NH, (long long)S2 * 3 * D, HD, (long long)S2 * 3 * D, HD,
         (long long)NH * S2 * S2, (long long)S2 * S2, 0.17677669529663687f);
    k_softmax<<<nblk(Bz * NH * S2), 256, 0, stream>>>(scores, Bz * NH * S2, S2);
    // ctx[b,q,h*32+d] = attn @ V  (B strided: V(k,n) stride 768)
    gemm(scores, qkv + 2 * D, nullptr, ctx, S2, HD, S2, S2, 1, 3 * D, D,
         Bz * NH, NH, (long long)NH * S2 * S2, (long long)S2 * S2,
         (long long)S2 * 3 * D, HD, (long long)S2 * D, HD, 1.f);
    gemm1(ctx, e.Wo, e.bo, hb, Mx, D, D);
    k_add<<<nblk((long long)Mx * D), 256, 0, stream>>>(xb, hb, (long long)Mx * D);
    k_ln<<<nblk(Mx), 256, 0, stream>>>(xb, Mx, D, e.ln2g, e.ln2b, hb);
    gemm1(hb, e.W1, e.b1, ffb, Mx, 512, D);
    k_gelu<<<nblk((long long)Mx * 512), 256, 0, stream>>>(ffb, (long long)Mx * 512);
    gemm1(ffb, e.W2, e.b2, hb, Mx, D, 512);
    k_add<<<nblk((long long)Mx * D), 256, 0, stream>>>(xb, hb, (long long)Mx * D);
  }
  {
    float* p0 = run_mlp(xb, Mx, &l2post, 1, bufA, bufB);  // -> bufA
    k_add<<<nblk((long long)Mx * D), 256, 0, stream>>>(l2_pts, p0, (long long)Mx * D);
  }

  // ---- SA3 (global) ----
  k_copycols<<<nblk((long long)Mx * 3), 256, 0, stream>>>(bufB, 259, 0, nxyz2, 3, 1, 3, (long long)Mx * 3);
  k_copycols<<<nblk((long long)Mx * 256), 256, 0, stream>>>(bufB, 259, 3, l2_pts, 256, 1, 256, (long long)Mx * 256);
  {
    float* o3 = run_mlp(bufB, Mx, sa3, 3, bufA, bufB);  // -> bufA [1024,1024]
    long long tot = (long long)Bz * 1024;
    k_maxk<<<nblk(tot), 256, 0, stream>>>(o3, S2, 1024, l3_pts, tot);
  }

  // ---- FP3: broadcast global feature, MLP -> l2f (bufB) ----
  k_copycols<<<nblk((long long)Mx * 256), 256, 0, stream>>>(bufB, 1280, 0, l2_pts, 256, 1, 256, (long long)Mx * 256);
  k_copycols<<<nblk((long long)Mx * 1024), 256, 0, stream>>>(bufB, 1280, 256, l3_pts, 1024, S2, 1024, (long long)Mx * 1024);
  float* l2f = run_mlp(bufB, Mx, fp3p, 2, bufA, bufB);  // -> bufB [1024,256]

  // ---- FP2: 3-NN interp l2f -> l1 resolution ----
  k_nn3<<<nblk(Bz * S1), 256, 0, stream>>>(nxyz1, S1, nxyz2, S2, Bz, idx3a, w3a);
  k_copycols<<<nblk((long long)Bz * S1 * 128), 256, 0, stream>>>(bufA, 384, 0, l1_pts, 128, 1, 128, (long long)Bz * S1 * 128);
  k_interp<<<nblk((long long)Bz * S1 * 256), 256, 0, stream>>>(bufA, 384, 128, l2f, 256, S2, idx3a, w3a, S1, (long long)Bz * S1 * 256);
  float* l1f = run_mlp(bufA, Bz * S1, fp2p, 2, bufB, bufA);  // -> bufA [4096,128]

  // ---- FP1: 3-NN interp l1f -> full resolution ----
  k_nn3<<<nblk((long long)Bz * Np), 256, 0, stream>>>(xyz0, Np, nxyz1, S1, Bz, idx3b, w3b);
  k_copycols<<<nblk((long long)Bz * Np * 6), 256, 0, stream>>>(bufB, 134, 0, pts0, 6, 1, 6, (long long)Bz * Np * 6);
  k_interp<<<nblk((long long)Bz * Np * 128), 256, 0, stream>>>(bufB, 134, 6, l1f, 128, S1, idx3b, w3b, Np, (long long)Bz * Np * 128);
  float* l0f = run_mlp(bufB, Bz * Np, fp1p, 3, bufA, bufB);  // -> bufA [65536,128]

  // ---- shared MLP + heads ----
  float* feat = run_mlp(l0f, Bz * Np, &sharedL, 1, bufB, bufA);  // -> bufB
  k_head<<<nblk((long long)Bz * 5 * Np), 256, 0, stream>>>(feat, Np, 128, semW, semb, 5,
                                                           (float*)d_out, Bz * 5 * Np);
  k_head<<<nblk((long long)Bz * 2 * Np), 256, 0, stream>>>(feat, Np, 128, defW, defb, 2,
                                                           (float*)d_out + (size_t)Bz * 5 * Np, Bz * 2 * Np);
}
